// HybridMoEModel_84679575208535
// MI455X (gfx1250) — matrix operand order repaired and checked by reading, not compiled, not run on hardware
//
#include <hip/hip_runtime.h>
#include <hip/hip_bf16.h>
#include <math.h>

typedef __attribute__((ext_vector_type(16))) _Float16 v16h;
typedef __attribute__((ext_vector_type(8)))  float    v8f;

union Frag {                // 16 halves = 32B = two ds_load_b128
    uint4 u[2];
    v16h  v;
};

// CDNA5 async global->LDS copy (no VGPR round-trip, tracked by ASYNCcnt).
// lds: 32-bit LDS byte address (low 32 bits of generic LDS pointer),
// g:   64-bit global address.  16B per lane per issue.
#define ASYNC_CP16(lds, g, imm_off)                                         \
    asm volatile("global_load_async_to_lds_b128 %0, %1, off offset:" #imm_off \
                 :: "v"(lds), "v"(g) : "memory")
#define ASYNC_WAIT() asm volatile("s_wait_asynccnt 0x0" ::: "memory")

static __device__ __forceinline__ unsigned ldsAddr(const void* p)
{
    return (unsigned)(uintptr_t)p;   // generic LDS addr low 32 bits == LDS offset
}

// ---------------- model constants ----------------
constexpr int D_MODEL    = 2048;
constexpr int N_HEADS    = 16;
constexpr int N_KV_HEADS = 4;
constexpr int HEAD_DIM   = 128;
constexpr int N_REP      = 4;
constexpr int N_EXPERTS  = 8;
constexpr int INTER      = 1024;
constexpr float EPS      = 1e-5f;
constexpr float AUX_COEFF = 0.01f;
constexpr int B_ = 2, L_ = 2048;
constexpr int T_ = B_ * L_;          // 4096 tokens

// =====================================================================
// RMSNorm: fp32 in -> f16 out (one block per token row)
// =====================================================================
__global__ __launch_bounds__(256)
void rmsnorm_kernel(const float* __restrict__ x, const float* __restrict__ w,
                    _Float16* __restrict__ out)
{
    __shared__ float red[256];
    const int row = blockIdx.x;
    const float* xr = x + (size_t)row * D_MODEL;
    float acc = 0.f;
    for (int i = threadIdx.x; i < D_MODEL; i += 256) { float v = xr[i]; acc += v * v; }
    red[threadIdx.x] = acc;
    __syncthreads();
    for (int s = 128; s > 0; s >>= 1) {
        if (threadIdx.x < s) red[threadIdx.x] += red[threadIdx.x + s];
        __syncthreads();
    }
    const float r = rsqrtf(red[0] / (float)D_MODEL + EPS);
    for (int i = threadIdx.x; i < D_MODEL; i += 256)
        out[(size_t)row * D_MODEL + i] = (_Float16)(xr[i] * r * w[i]);
}

// =====================================================================
// Tiled WMMA GEMM: C(MxN) = A_f16(MxK) * B_f32->f16(KxN), N/K compile-time.
// 128x64 tile / block, BK=64, 16 waves; each wave: two stacked 16x16 C tiles
// sharing one B fragment.  A tile streamed global->LDS with ASYNC loads;
// B tile converted in-register (K-strided reads fold into imm offsets) and
// stored transposed so all WMMA fragments are contiguous 2x ds_load_b128.
// MODE 0: f16 out.  MODE 1: fp32 out = acc + resid.  MODE 2: fp32 out += wte*acc.
// =====================================================================
template<int MODE, int N, int K>
__global__ __launch_bounds__(512)
void gemm_wmma(const _Float16* __restrict__ A, const float* __restrict__ Bw,
               _Float16* __restrict__ outH, float* __restrict__ outF,
               const float* __restrict__ resid,
               const float* __restrict__ wte, int expert)
{
    __shared__ _Float16 As[128][88];    // row-major, pitch 176B (16B aligned)
    __shared__ _Float16 BsT[64][88];    // transposed: [n][k]
    __shared__ float    sw[128];
    __shared__ int      anyFlag;

    const int tid     = threadIdx.x;
    const int rowBase = blockIdx.y * 128;
    const int colBase = blockIdx.x * 64;

    if (MODE == 2) {   // skip the tile if no token in it routed to this expert
        if (tid == 0) anyFlag = 0;
        __syncthreads();
        if (tid < 128) {
            float w = wte[(size_t)(rowBase + tid) * N_EXPERTS + expert];
            sw[tid] = w;
            if (w != 0.f) anyFlag = 1;
        }
        __syncthreads();
        if (!anyFlag) return;
    }

    const int lane = tid & 31, wave = tid >> 5;   // wave32
    const int wmA  = wave >> 2, wn = wave & 3;
    const int ml   = lane & 15, hh = lane >> 4;

    const int ar = tid >> 2;            // A loader: row 0..127
    const int ac = (tid & 3) * 16;      //           col 0/16/32/48 (2x16B)
    const int bn = tid & 63;            // B loader: n
    const int bk = (tid >> 6) * 8;      //           k base (8 strided floats)

    const unsigned ldsA = ldsAddr(&As[ar][ac]);
    const _Float16* gA  = A + (size_t)(rowBase + ar) * K + ac;
    const float*    gB  = Bw + (size_t)bk * N + colBase + bn;

    v8f c0 = {}, c1 = {};

    for (int kt = 0; kt < K; kt += 64) {
        // A tile 128x64 halves: async global->LDS, 2x16B per thread
        ASYNC_CP16(ldsA, gA + kt, 0);
        ASYNC_CP16(ldsA, gA + kt, 16);
        {   // B tile 64x64 fp32 -> f16, K-strided reads (imm offsets), transposed store
            const float* src = gB + (size_t)kt * N;
            alignas(16) _Float16 h8[8];
            #pragma unroll
            for (int i = 0; i < 8; ++i) h8[i] = (_Float16)src[(size_t)i * N];
            *reinterpret_cast<uint4*>(&BsT[bn][bk]) = *reinterpret_cast<const uint4*>(h8);
        }
        if (kt + 64 < K && (bn & 31) == 0)   // prefetch next weight tile
            __builtin_prefetch(gB + (size_t)(kt + 64) * N, 0, 3);
        ASYNC_WAIT();
        __syncthreads();
        #pragma unroll
        for (int k0 = 0; k0 < 64; k0 += 32) {
            Frag a0, a1, b;
            a0.u[0] = *reinterpret_cast<const uint4*>(&As[wmA*16 + ml][k0 + hh*8]);
            a0.u[1] = *reinterpret_cast<const uint4*>(&As[wmA*16 + ml][k0 + 16 + hh*8]);
            a1.u[0] = *reinterpret_cast<const uint4*>(&As[(wmA+4)*16 + ml][k0 + hh*8]);
            a1.u[1] = *reinterpret_cast<const uint4*>(&As[(wmA+4)*16 + ml][k0 + 16 + hh*8]);
            b.u[0]  = *reinterpret_cast<const uint4*>(&BsT[wn*16 + ml][k0 + hh*8]);
            b.u[1]  = *reinterpret_cast<const uint4*>(&BsT[wn*16 + ml][k0 + 16 + hh*8]);
            c0 = __builtin_amdgcn_wmma_f32_16x16x32_f16(false, a0.v, false, b.v, (short)0, c0, false, false);
            c1 = __builtin_amdgcn_wmma_f32_16x16x32_f16(false, a1.v, false, b.v, (short)0, c1, false, false);
        }
        __syncthreads();
    }

    #pragma unroll
    for (int r = 0; r < 8; ++r) {
        const int m = (hh << 3) + r;
        const int gm0 = rowBase + wmA*16 + m;
        const int gm1 = gm0 + 64;
        const size_t gn = colBase + wn*16 + ml;
        const size_t o0 = (size_t)gm0 * N + gn;
        const size_t o1 = (size_t)gm1 * N + gn;
        if (MODE == 0)      { outH[o0] = (_Float16)c0[r];      outH[o1] = (_Float16)c1[r]; }
        else if (MODE == 1) { outF[o0] = c0[r] + resid[o0];    outF[o1] = c1[r] + resid[o1]; }
        else { outF[o0] += sw[wmA*16 + m] * c0[r];  outF[o1] += sw[wmA*16 + 64 + m] * c1[r]; }
    }
}

// =====================================================================
// Dual-B WMMA GEMM for MoE up-proj: out = silu(A*B1) * (A*B2), f16 out
// =====================================================================
template<int N, int K>
__global__ __launch_bounds__(512)
void gemm_dual_silu(const _Float16* __restrict__ A,
                    const float* __restrict__ B1w, const float* __restrict__ B2w,
                    _Float16* __restrict__ outH)
{
    __shared__ _Float16 As  [128][88];
    __shared__ _Float16 B1sT[64][88];
    __shared__ _Float16 B2sT[64][88];

    const int tid     = threadIdx.x;
    const int rowBase = blockIdx.y * 128;
    const int colBase = blockIdx.x * 64;
    const int lane = tid & 31, wave = tid >> 5;
    const int wmA = wave >> 2, wn = wave & 3;
    const int ml = lane & 15, hh = lane >> 4;
    const int ar = tid >> 2, ac = (tid & 3) * 16;
    const int bn = tid & 63, bk = (tid >> 6) * 8;

    const unsigned ldsA = ldsAddr(&As[ar][ac]);
    const _Float16* gA  = A + (size_t)(rowBase + ar) * K + ac;
    const float*    gB1 = B1w + (size_t)bk * N + colBase + bn;
    const float*    gB2 = B2w + (size_t)bk * N + colBase + bn;

    v8f c10 = {}, c11 = {}, c20 = {}, c21 = {};

    for (int kt = 0; kt < K; kt += 64) {
        ASYNC_CP16(ldsA, gA + kt, 0);
        ASYNC_CP16(ldsA, gA + kt, 16);
        {
            const float* s1 = gB1 + (size_t)kt * N;
            const float* s2 = gB2 + (size_t)kt * N;
            alignas(16) _Float16 h1[8], h2[8];
            #pragma unroll
            for (int i = 0; i < 8; ++i) {
                h1[i] = (_Float16)s1[(size_t)i * N];
                h2[i] = (_Float16)s2[(size_t)i * N];
            }
            *reinterpret_cast<uint4*>(&B1sT[bn][bk]) = *reinterpret_cast<const uint4*>(h1);
            *reinterpret_cast<uint4*>(&B2sT[bn][bk]) = *reinterpret_cast<const uint4*>(h2);
        }
        if (kt + 64 < K && (bn & 31) == 0) {
            __builtin_prefetch(gB1 + (size_t)(kt + 64) * N, 0, 3);
            __builtin_prefetch(gB2 + (size_t)(kt + 64) * N, 0, 3);
        }
        ASYNC_WAIT();
        __syncthreads();
        #pragma unroll
        for (int k0 = 0; k0 < 64; k0 += 32) {
            Frag a0, a1, b1, b2;
            a0.u[0] = *reinterpret_cast<const uint4*>(&As[wmA*16 + ml][k0 + hh*8]);
            a0.u[1] = *reinterpret_cast<const uint4*>(&As[wmA*16 + ml][k0 + 16 + hh*8]);
            a1.u[0] = *reinterpret_cast<const uint4*>(&As[(wmA+4)*16 + ml][k0 + hh*8]);
            a1.u[1] = *reinterpret_cast<const uint4*>(&As[(wmA+4)*16 + ml][k0 + 16 + hh*8]);
            b1.u[0] = *reinterpret_cast<const uint4*>(&B1sT[wn*16 + ml][k0 + hh*8]);
            b1.u[1] = *reinterpret_cast<const uint4*>(&B1sT[wn*16 + ml][k0 + 16 + hh*8]);
            b2.u[0] = *reinterpret_cast<const uint4*>(&B2sT[wn*16 + ml][k0 + hh*8]);
            b2.u[1] = *reinterpret_cast<const uint4*>(&B2sT[wn*16 + ml][k0 + 16 + hh*8]);
            c10 = __builtin_amdgcn_wmma_f32_16x16x32_f16(false, a0.v, false, b1.v, (short)0, c10, false, false);
            c11 = __builtin_amdgcn_wmma_f32_16x16x32_f16(false, a1.v, false, b1.v, (short)0, c11, false, false);
            c20 = __builtin_amdgcn_wmma_f32_16x16x32_f16(false, a0.v, false, b2.v, (short)0, c20, false, false);
            c21 = __builtin_amdgcn_wmma_f32_16x16x32_f16(false, a1.v, false, b2.v, (short)0, c21, false, false);
        }
        __syncthreads();
    }

    #pragma unroll
    for (int r = 0; r < 8; ++r) {
        const int m = (hh << 3) + r;
        const size_t gn = colBase + wn*16 + ml;
        const size_t o0 = (size_t)(rowBase + wmA*16 + m) * N + gn;
        const size_t o1 = (size_t)(rowBase + wmA*16 + 64 + m) * N + gn;
        const float g0 = c10[r], g1 = c11[r];
        const float s0 = g0 * __builtin_amdgcn_rcpf(1.f + __expf(-g0));
        const float s1 = g1 * __builtin_amdgcn_rcpf(1.f + __expf(-g1));
        outH[o0] = (_Float16)(s0 * c20[r]);
        outH[o1] = (_Float16)(s1 * c21[r]);
    }
}

// =====================================================================
// In-place RoPE on f16 (b,l,nh,128) buffer; one thread per (pos,head,pair)
// =====================================================================
__global__ void rope_kernel(_Float16* __restrict__ buf, int nHeads)
{
    const size_t idx = (size_t)blockIdx.x * blockDim.x + threadIdx.x;
    const int j = (int)(idx & 63);
    const size_t t = idx >> 6;                 // (b*L + l)*nHeads + h
    const int l = (int)((t / nHeads) % L_);
    _Float16* p = buf + t * HEAD_DIM + 2 * j;
    const float inv = __expf(-(float)j * (logf(10000.f) / 64.f));
    const float ang = (float)l * inv;
    float s, c;
    __sincosf(ang, &s, &c);                    // v_sin_f32 / v_cos_f32
    const float xr = (float)p[0], xi = (float)p[1];
    p[0] = (_Float16)(xr * c - xi * s);
    p[1] = (_Float16)(xr * s + xi * c);
}

// =====================================================================
// Flash attention: block = (16 q-rows, head, batch), 8 waves / 256 threads.
// QK^T and P*V on WMMA; Q/K tiles streamed with ASYNC global->LDS loads;
// V transposed through registers.
// =====================================================================
__global__ __launch_bounds__(256)
void attn_kernel(const _Float16* __restrict__ q, const _Float16* __restrict__ k,
                 const _Float16* __restrict__ v, _Float16* __restrict__ ao)
{
    __shared__ _Float16 Qs[16][136];     // pitch 272B: 68 dwords -> clean bank stepping
    __shared__ _Float16 Ks[64][136];
    __shared__ _Float16 VsT[128][88];    // transposed: [d][key]
    __shared__ float    Ss[16][64];
    __shared__ _Float16 Ps[16][88];
    __shared__ float    mS[16], lS[16], aS[16];

    const int qt = blockIdx.x, h = blockIdx.y, b = blockIdx.z;
    const int kv = h / N_REP;
    const int qBase = qt * 16;
    const int tid = threadIdx.x;
    const int lane = tid & 31, wave = tid >> 5;
    const int ml = lane & 15, hh = lane >> 4;
    const float scale = 0.08838834764831845f;   // 1/sqrt(128)

    {   // Q tile 16x128: async copy, 16B per thread
        const int r = tid >> 4, cc = (tid & 15) * 8;
        const unsigned ldsQ = ldsAddr(&Qs[r][cc]);
        const _Float16* gQ =
            q + (((size_t)(b * L_ + qBase + r) * N_HEADS) + h) * HEAD_DIM + cc;
        ASYNC_CP16(ldsQ, gQ, 0);
    }
    if (tid < 16) { mS[tid] = -1e30f; lS[tid] = 0.f; }

    v8f o = {};
    const int ktMax = (qBase + 15) / 64;

    const int kr = tid >> 2, kc = (tid & 3) * 32;   // K/V loader mapping
    const unsigned ldsK = ldsAddr(&Ks[kr][kc]);

    for (int kt = 0; kt <= ktMax; ++kt) {
        __syncthreads();                       // previous-iter tiles consumed
        {   // K tile 64x128: async copy (4x16B); V tile transposed via registers
            const _Float16* gK = k + (((size_t)(b * L_ + kt * 64 + kr) * N_KV_HEADS) + kv) * HEAD_DIM + kc;
            ASYNC_CP16(ldsK, gK, 0);
            ASYNC_CP16(ldsK, gK, 16);
            ASYNC_CP16(ldsK, gK, 32);
            ASYNC_CP16(ldsK, gK, 48);
            const _Float16* vs = v + (((size_t)(b * L_ + kt * 64 + kr) * N_KV_HEADS) + kv) * HEAD_DIM + kc;
            union { uint4 u[4]; _Float16 hh16[32]; } vl;
            #pragma unroll
            for (int i = 0; i < 4; ++i) vl.u[i] = reinterpret_cast<const uint4*>(vs)[i];
            #pragma unroll
            for (int i = 0; i < 32; ++i) VsT[kc + i][kr] = vl.hh16[i];
        }
        ASYNC_WAIT();
        __syncthreads();
        if (wave < 4) {                        // S(16x64) = Q * K^T, one 16x16 tile per wave
            v8f s = {};
            #pragma unroll
            for (int k0 = 0; k0 < 128; k0 += 32) {
                Frag a, bb;
                a.u[0]  = *reinterpret_cast<const uint4*>(&Qs[ml][k0 + hh*8]);
                a.u[1]  = *reinterpret_cast<const uint4*>(&Qs[ml][k0 + 16 + hh*8]);
                bb.u[0] = *reinterpret_cast<const uint4*>(&Ks[wave*16 + ml][k0 + hh*8]);
                bb.u[1] = *reinterpret_cast<const uint4*>(&Ks[wave*16 + ml][k0 + 16 + hh*8]);
                s = __builtin_amdgcn_wmma_f32_16x16x32_f16(false, a.v, false, bb.v, (short)0, s, false, false);
            }
            #pragma unroll
            for (int r = 0; r < 8; ++r) {
                const int m = (hh << 3) + r;
                const int n = wave * 16 + ml;
                float val = s[r] * scale;
                if (kt * 64 + n > qBase + m) val = -1e30f;    // causal mask
                Ss[m][n] = val;
            }
        }
        __syncthreads();
        if (tid < 16) {                        // online softmax, one lane per row
            const int m = tid;
            const float mOld = mS[m];
            float mx = mOld;
            for (int n = 0; n < 64; ++n) mx = fmaxf(mx, Ss[m][n]);
            const float alpha = __expf(mOld - mx);
            float sum = alpha * lS[m];
            for (int n = 0; n < 64; ++n) {
                const float p = __expf(Ss[m][n] - mx);
                Ps[m][n] = (_Float16)p;
                sum += p;
            }
            mS[m] = mx; lS[m] = sum; aS[m] = alpha;
        }
        __syncthreads();
        {   // O(16x128) = alpha*O + P * V, one 16x16 col-tile per wave
            #pragma unroll
            for (int r = 0; r < 8; ++r) o[r] *= aS[(hh << 3) + r];
            #pragma unroll
            for (int k0 = 0; k0 < 64; k0 += 32) {
                Frag a, bb;
                a.u[0]  = *reinterpret_cast<const uint4*>(&Ps[ml][k0 + hh*8]);
                a.u[1]  = *reinterpret_cast<const uint4*>(&Ps[ml][k0 + 16 + hh*8]);
                bb.u[0] = *reinterpret_cast<const uint4*>(&VsT[wave*16 + ml][k0 + hh*8]);
                bb.u[1] = *reinterpret_cast<const uint4*>(&VsT[wave*16 + ml][k0 + 16 + hh*8]);
                o = __builtin_amdgcn_wmma_f32_16x16x32_f16(false, a.v, false, bb.v, (short)0, o, false, false);
            }
        }
    }
    __syncthreads();
    #pragma unroll
    for (int r = 0; r < 8; ++r) {
        const int m = (hh << 3) + r;
        ao[(((size_t)(b * L_ + qBase + m) * N_HEADS) + h) * HEAD_DIM + wave * 16 + ml] =
            (_Float16)(o[r] / lS[m]);
    }
}

// =====================================================================
// MoE gating: softmax over 8, top-2 normalized weights, aux-loss accumulators
// =====================================================================
__global__ __launch_bounds__(64)
void gate_kernel(const _Float16* __restrict__ hm, const float* __restrict__ gw,
                 float* __restrict__ wte, float* __restrict__ auxAcc)
{
    __shared__ float red[64][8];
    const int t = blockIdx.x;
    float part[8] = {};
    for (int d = threadIdx.x; d < D_MODEL; d += 64) {
        const float xv = (float)hm[(size_t)t * D_MODEL + d];
        const float* g = gw + (size_t)d * N_EXPERTS;
        #pragma unroll
        for (int e = 0; e < 8; ++e) part[e] += xv * g[e];
    }
    #pragma unroll
    for (int e = 0; e < 8; ++e) red[threadIdx.x][e] = part[e];
    __syncthreads();
    if (threadIdx.x == 0) {
        float logit[8];
        for (int e = 0; e < 8; ++e) {
            float s = 0.f;
            for (int i = 0; i < 64; ++i) s += red[i][e];
            logit[e] = s;
        }
        float mx = logit[0];
        for (int e = 1; e < 8; ++e) mx = fmaxf(mx, logit[e]);
        float se = 0.f, sc[8];
        for (int e = 0; e < 8; ++e) { sc[e] = __expf(logit[e] - mx); se += sc[e]; }
        for (int e = 0; e < 8; ++e) sc[e] /= se;
        int i0 = 0;
        for (int e = 1; e < 8; ++e) if (sc[e] > sc[i0]) i0 = e;
        int i1 = (i0 == 0) ? 1 : 0;
        for (int e = 0; e < 8; ++e) if (e != i0 && sc[e] > sc[i1]) i1 = e;
        const float denom = sc[i0] + sc[i1];
        for (int e = 0; e < 8; ++e) wte[(size_t)t * 8 + e] = 0.f;
        wte[(size_t)t * 8 + i0] = sc[i0] / denom;
        wte[(size_t)t * 8 + i1] = sc[i1] / denom;
        for (int e = 0; e < 8; ++e) atomicAdd(&auxAcc[e], sc[e]);  // p sums
        atomicAdd(&auxAcc[8 + i0], 1.f);                           // f counts
        atomicAdd(&auxAcc[8 + i1], 1.f);
    }
}

__global__ void zero_aux(float* auxAcc)
{
    if (threadIdx.x < 16) auxAcc[threadIdx.x] = 0.f;
}

__global__ void aux_finalize(const float* __restrict__ auxAcc, float* __restrict__ out)
{
    if (threadIdx.x == 0) {
        float s = 0.f;
        for (int e = 0; e < 8; ++e)
            s += (auxAcc[8 + e] / (float)T_) * (auxAcc[e] / (float)T_);
        out[0] = AUX_COEFF * (float)N_EXPERTS * s;
    }
}

// =====================================================================
extern "C" void kernel_launch(void* const* d_in, const int* in_sizes, int n_in,
                              void* d_out, int out_size, void* d_ws, size_t ws_size,
                              hipStream_t stream)
{
    (void)in_sizes; (void)n_in; (void)out_size; (void)ws_size;
    const float* x   = (const float*)d_in[0];
    const float* w_q = (const float*)d_in[1];
    const float* w_k = (const float*)d_in[2];
    const float* w_v = (const float*)d_in[3];
    const float* w_o = (const float*)d_in[4];
    const float* anw = (const float*)d_in[5];
    const float* mnw = (const float*)d_in[6];
    const float* gw  = (const float*)d_in[7];
    const float* w1  = (const float*)d_in[8];
    const float* w3  = (const float*)d_in[9];
    const float* w2  = (const float*)d_in[10];
    float* out = (float*)d_out;

    char* ws = (char*)d_ws;
    size_t off = 0;
    auto alloc = [&](size_t bytes) -> void* {
        void* p = ws + off;
        off = (off + bytes + 255) & ~(size_t)255;
        return p;
    };
    _Float16* h    = (_Float16*)alloc((size_t)T_ * D_MODEL * 2);
    _Float16* qb   = (_Float16*)alloc((size_t)T_ * D_MODEL * 2);
    _Float16* kb   = (_Float16*)alloc((size_t)T_ * N_KV_HEADS * HEAD_DIM * 2);
    _Float16* vb   = (_Float16*)alloc((size_t)T_ * N_KV_HEADS * HEAD_DIM * 2);
    _Float16* aob  = (_Float16*)alloc((size_t)T_ * D_MODEL * 2);
    _Float16* hm   = (_Float16*)alloc((size_t)T_ * D_MODEL * 2);
    _Float16* act  = (_Float16*)alloc((size_t)T_ * INTER * 2);
    float*    wte  = (float*)alloc((size_t)T_ * N_EXPERTS * 4);
    float*    aux  = (float*)alloc(64 * 4);

    zero_aux<<<1, 32, 0, stream>>>(aux);

    // --- attention block ---
    rmsnorm_kernel<<<T_, 256, 0, stream>>>(x, anw, h);
    gemm_wmma<0, D_MODEL, D_MODEL><<<dim3(D_MODEL/64, T_/128), 512, 0, stream>>>(
        h, w_q, qb, nullptr, nullptr, nullptr, 0);
    gemm_wmma<0, N_KV_HEADS*HEAD_DIM, D_MODEL><<<dim3((N_KV_HEADS*HEAD_DIM)/64, T_/128), 512, 0, stream>>>(
        h, w_k, kb, nullptr, nullptr, nullptr, 0);
    gemm_wmma<0, N_KV_HEADS*HEAD_DIM, D_MODEL><<<dim3((N_KV_HEADS*HEAD_DIM)/64, T_/128), 512, 0, stream>>>(
        h, w_v, vb, nullptr, nullptr, nullptr, 0);
    rope_kernel<<<(T_ * N_HEADS * 64) / 256, 256, 0, stream>>>(qb, N_HEADS);
    rope_kernel<<<(T_ * N_KV_HEADS * 64) / 256, 256, 0, stream>>>(kb, N_KV_HEADS);
    attn_kernel<<<dim3(L_/16, N_HEADS, B_), 256, 0, stream>>>(qb, kb, vb, aob);
    gemm_wmma<1, D_MODEL, D_MODEL><<<dim3(D_MODEL/64, T_/128), 512, 0, stream>>>(
        aob, w_o, nullptr, out, x, nullptr, 0);  // x1 -> d_out

    // --- MoE block ---
    rmsnorm_kernel<<<T_, 256, 0, stream>>>(out, mnw, hm);
    gate_kernel<<<T_, 64, 0, stream>>>(hm, gw, wte, aux);
    for (int e = 0; e < N_EXPERTS; ++e) {
        gemm_dual_silu<INTER, D_MODEL><<<dim3(INTER/64, T_/128), 512, 0, stream>>>(
            hm, w1 + (size_t)e * D_MODEL * INTER, w3 + (size_t)e * D_MODEL * INTER, act);
        gemm_wmma<2, D_MODEL, INTER><<<dim3(D_MODEL/64, T_/128), 512, 0, stream>>>(
            act, w2 + (size_t)e * INTER * D_MODEL, nullptr, out, nullptr, wte, e);
    }
    aux_finalize<<<1, 32, 0, stream>>>(aux, out + (size_t)T_ * D_MODEL);
}
